// AttentionDecoder_78529182040729
// MI455X (gfx1250) — matrix operand order repaired and checked by reading, not compiled, not run on hardware
//
#include <hip/hip_runtime.h>
#include <hip/hip_bf16.h>

// MI455X / gfx1250, wave32. GEMMs on v_wmma_f32_16x16x32_bf16 with bf16-staged
// operands; 32x64 register blocking per wave (8 accumulators, branch-free loop).

typedef __attribute__((ext_vector_type(16))) __bf16 v16bf;
typedef __attribute__((ext_vector_type(8)))  __bf16 v8bf;
typedef __attribute__((ext_vector_type(8)))  float  v8f;

#define B_   32
#define T_   512
#define E_   512
#define H_   512
#define H3_  1536
#define L_   200
#define C_   5000
#define SOS_ 1

static __device__ __forceinline__ v8f wmma_bf16(v16bf a, v16bf b, v8f c) {
  // (neg_a, A, neg_b, B, c_mod, C, reuse_a, reuse_b)
  return __builtin_amdgcn_wmma_f32_16x16x32_bf16(false, a, false, b, (short)0, c,
                                                 false, false);
}

// A fragment 16x32 bf16 (ISA 7.12.2): lane<16 -> M=lane, K=k0+{0..7,16..23};
// lane>=16 -> M=lane-16, K=k0+{8..15,24..31}.  Two contiguous 16B loads.
static __device__ __forceinline__ v16bf load_a_bf(const __bf16* A, int lda, int row0,
                                                  int k0) {
  int lane = threadIdx.x & 31;
  const __bf16* p =
      A + (size_t)(row0 + (lane & 15)) * lda + (k0 + ((lane >> 4) << 3));
  v8bf lo = *(const v8bf*)p;
  v8bf hi = *(const v8bf*)(p + 16);
  return __builtin_shufflevector(lo, hi, 0, 1, 2, 3, 4, 5, 6, 7, 8, 9, 10, 11, 12,
                                 13, 14, 15);
}

// Same layout, pre-resolved (gathered) row pointer.
static __device__ __forceinline__ v16bf load_a_row(const __bf16* Arow, int k0) {
  v8bf lo = *(const v8bf*)(Arow + k0);
  v8bf hi = *(const v8bf*)(Arow + k0 + 16);
  return __builtin_shufflevector(lo, hi, 0, 1, 2, 3, 4, 5, 6, 7, 8, 9, 10, 11, 12,
                                 13, 14, 15);
}

// B fragment 32x16 (KxN) from row-major bf16 W[N][K]: two contiguous 16B loads.
// lanes 0-15: N=lane, K=k0+{0..7,16..23} ; lanes 16-31: N=lane-16, K=k0+{8..15,24..31}.
// NOTE: branch-free. Rows n >= nmax are clamped to row nmax-1;
// out-of-range columns are simply never stored.
static __device__ __forceinline__ v16bf load_b_bf(const __bf16* W, int ldw, int col0,
                                                  int k0, int nmax) {
  int lane = threadIdx.x & 31;
  int n = col0 + (lane & 15);
  n = (n < nmax) ? n : (nmax - 1);  // clamp: valid load, dead columns unstored
  const __bf16* p = W + (size_t)n * ldw + (k0 + ((lane >> 4) << 3));
  v8bf lo = *(const v8bf*)p;
  v8bf hi = *(const v8bf*)(p + 16);
  return __builtin_shufflevector(lo, hi, 0, 1, 2, 3, 4, 5, 6, 7, 8, 9, 10, 11, 12,
                                 13, 14, 15);
}

// C/D tile store (8 VGPRs): M = r + 8*(lane>=16), N = lane&15.
static __device__ __forceinline__ void store_tile(float* C, int ldc, int row0,
                                                  int col0, int nmax, v8f acc,
                                                  const float* bias) {
  int lane = threadIdx.x & 31;
  int n = col0 + (lane & 15);
  if (n >= nmax) return;
  float add = bias ? bias[n] : 0.f;
  int mb = row0 + ((lane >> 4) << 3);
#pragma unroll
  for (int r = 0; r < 8; ++r) C[(size_t)(mb + r) * ldc + n] = acc[r] + add;
}

// ---- conversion / init kernels -------------------------------------------

__global__ void conv_kernel(const float* __restrict__ src, __bf16* __restrict__ dst,
                            int n) {
  int i = (blockIdx.x * blockDim.x + threadIdx.x) * 2;
  if (i < n) {  // pairs -> v_cvt_pk_bf16_f32 + b32 store
    dst[i] = (__bf16)src[i];
    dst[i + 1] = (__bf16)src[i + 1];
  }
}

// split W_ih[3H, H+E] into compact bf16 Wx[3H,H] and Wc[3H,E]
__global__ void conv_wih_kernel(const float* __restrict__ W_ih,
                                __bf16* __restrict__ wx, __bf16* __restrict__ wc) {
  int i = blockIdx.x * blockDim.x + threadIdx.x;  // 3H * (H+E)
  int r = i >> 10, c = i & 1023;
  float v = W_ih[i];
  if (c < H_) wx[(size_t)r * H_ + c] = (__bf16)v;
  else        wc[(size_t)r * E_ + (c - H_)] = (__bf16)v;
}

__global__ void init_h_kernel(float* __restrict__ h, __bf16* __restrict__ h_bf) {
  int i = blockIdx.x * blockDim.x + threadIdx.x;  // B*H
  h[i] = 0.f;
  h_bf[i] = (__bf16)0.f;
}

// ---- big GEMMs: each wave owns a 32x64 strip (2 A-frags x 4 B-frags = 8 WMMA)

// proj[B*T, H] = enc @ W_attn^T   (fp32 output kept for attention precision)
__global__ void proj_kernel(const __bf16* __restrict__ enc_bf,
                            const __bf16* __restrict__ wattn_bf,
                            float* __restrict__ proj) {
  int wave = threadIdx.x >> 5;
  int j = (blockIdx.x & 63) * 8 + wave;  // 512 wave-jobs over M
  int row0 = j * 32;
  int c0 = (blockIdx.x >> 6) * 64;       // 8 column groups
  v8f acc[2][4] = {};
  for (int k0 = 0; k0 < E_; k0 += 32) {
    v16bf aA = load_a_bf(enc_bf, E_, row0, k0);
    v16bf aB = load_a_bf(enc_bf, E_, row0 + 16, k0);
#pragma unroll
    for (int q = 0; q < 4; ++q) {
      v16bf b = load_b_bf(wattn_bf, E_, c0 + q * 16, k0, H_);
      acc[0][q] = wmma_bf16(aA, b, acc[0][q]);
      acc[1][q] = wmma_bf16(aB, b, acc[1][q]);
    }
  }
#pragma unroll
  for (int q = 0; q < 4; ++q) {
    store_tile(proj, H_, row0, c0 + q * 16, H_, acc[0][q], nullptr);
    store_tile(proj, H_, row0 + 16, c0 + q * 16, H_, acc[1][q], nullptr);
  }
}

// xgi[B*L, 3H] = emb_table[input_ids] @ W_ih[:, :H]^T + b_ih
__global__ void xgi_kernel(const int* __restrict__ targets,
                           const __bf16* __restrict__ emb_bf,
                           const __bf16* __restrict__ wihx_bf,
                           const float* __restrict__ b_ih,
                           float* __restrict__ xgi) {
  int wave = threadIdx.x >> 5;
  int lane = threadIdx.x & 31;
  int j = (blockIdx.x % 25) * 8 + wave;  // 200 wave-jobs over M (B*L/32)
  int row0 = j * 32;
  int c0 = (blockIdx.x / 25) * 64;       // 24 column groups
  // teacher-forcing gather for this lane's two A rows
  int rA = row0 + (lane & 15), rB = rA + 16;
  int bA = rA / L_, lA = rA - bA * L_;
  int bB = rB / L_, lB = rB - bB * L_;
  int idA = (lA == 0) ? SOS_ : targets[bA * L_ + lA - 1];
  int idB = (lB == 0) ? SOS_ : targets[bB * L_ + lB - 1];
  const __bf16* ArowA = emb_bf + (size_t)idA * H_ + ((lane >> 4) << 3);
  const __bf16* ArowB = emb_bf + (size_t)idB * H_ + ((lane >> 4) << 3);
  v8f acc[2][4] = {};
  for (int k0 = 0; k0 < H_; k0 += 32) {
    v16bf aA = load_a_row(ArowA, k0);
    v16bf aB = load_a_row(ArowB, k0);
#pragma unroll
    for (int q = 0; q < 4; ++q) {
      v16bf b = load_b_bf(wihx_bf, H_, c0 + q * 16, k0, H3_);
      acc[0][q] = wmma_bf16(aA, b, acc[0][q]);
      acc[1][q] = wmma_bf16(aB, b, acc[1][q]);
    }
  }
#pragma unroll
  for (int q = 0; q < 4; ++q) {
    store_tile(xgi, H3_, row0, c0 + q * 16, H3_, acc[0][q], b_ih);
    store_tile(xgi, H3_, row0 + 16, c0 + q * 16, H3_, acc[1][q], b_ih);
  }
}

// logits[B*L, C] = hs @ W_out^T + b_out : HBM-store-bound (128 MB out)
__global__ void logits_kernel(const __bf16* __restrict__ hs_bf,
                              const __bf16* __restrict__ wout_bf,
                              const float* __restrict__ b_out,
                              float* __restrict__ out) {
  int wave = threadIdx.x >> 5;
  int j = (blockIdx.x % 25) * 8 + wave;  // 200 wave-jobs over M
  int row0 = j * 32;
  int c0 = (blockIdx.x / 25) * 64;       // 79 column groups (last clamped)
  v8f acc[2][4] = {};
  for (int k0 = 0; k0 < H_; k0 += 32) {
    // speculative prefetch of next A chunk (global_prefetch_b8), branch-free
    __builtin_prefetch(hs_bf + (size_t)row0 * H_ + k0 + 32, 0, 0);
    v16bf aA = load_a_bf(hs_bf, H_, row0, k0);
    v16bf aB = load_a_bf(hs_bf, H_, row0 + 16, k0);
#pragma unroll
    for (int q = 0; q < 4; ++q) {
      v16bf b = load_b_bf(wout_bf, H_, c0 + q * 16, k0, C_);  // clamped rows
      acc[0][q] = wmma_bf16(aA, b, acc[0][q]);
      acc[1][q] = wmma_bf16(aB, b, acc[1][q]);
    }
  }
#pragma unroll
  for (int q = 0; q < 4; ++q) {
    store_tile(out, C_, row0, c0 + q * 16, C_, acc[0][q], b_out);
    store_tile(out, C_, row0 + 16, c0 + q * 16, C_, acc[1][q], b_out);
  }
}

// ---- recurrence ------------------------------------------------------------

// fused scores -> masked softmax -> context (bf16 out). One block per batch row.
// proj + enc (67 MB fp32) stay L2-resident across all 200 steps (192 MB L2).
__global__ void attn_kernel(const float* __restrict__ proj,
                            const float* __restrict__ enc,
                            const unsigned char* __restrict__ mask,
                            const float* __restrict__ h,
                            __bf16* __restrict__ ctx_bf) {
  __shared__ float sh_h[H_];
  __shared__ float sc[T_];
  __shared__ float red[16];
  int b = blockIdx.x;
  int tid = threadIdx.x;  // 256 threads = 8 waves
  int wave = tid >> 5, lane = tid & 31;
  for (int i = tid; i < H_; i += 256) sh_h[i] = h[b * H_ + i];
  __syncthreads();
  for (int t = wave; t < T_; t += 8) {
    const float* pe = proj + ((size_t)b * T_ + t) * H_;
    float acc = 0.f;
#pragma unroll 4
    for (int k = lane; k < H_; k += 32) acc += pe[k] * sh_h[k];
#pragma unroll
    for (int off = 16; off > 0; off >>= 1) acc += __shfl_xor(acc, off, 32);
    if (lane == 0) sc[t] = mask[b * T_ + t] ? acc : -1e30f;
  }
  __syncthreads();
  float lmax = -1e30f;
  for (int i = tid; i < T_; i += 256) lmax = fmaxf(lmax, sc[i]);
#pragma unroll
  for (int off = 16; off > 0; off >>= 1)
    lmax = fmaxf(lmax, __shfl_xor(lmax, off, 32));
  if (lane == 0) red[wave] = lmax;
  __syncthreads();
  if (tid == 0) {
    float v = red[0];
    for (int w = 1; w < 8; ++w) v = fmaxf(v, red[w]);
    red[8] = v;
  }
  __syncthreads();
  float gmax = red[8];
  float lsum = 0.f;
  for (int i = tid; i < T_; i += 256) {
    float e = __expf(sc[i] - gmax);
    sc[i] = e;
    lsum += e;
  }
#pragma unroll
  for (int off = 16; off > 0; off >>= 1) lsum += __shfl_xor(lsum, off, 32);
  if (lane == 0) red[wave] = lsum;
  __syncthreads();
  if (tid == 0) {
    float s = 0.f;
    for (int w = 0; w < 8; ++w) s += red[w];
    red[9] = s;
  }
  __syncthreads();
  float inv = 1.f / red[9];  // fold softmax normalization into context
  for (int e = tid; e < E_; e += 256) {
    float acc = 0.f;
    const float* eb = enc + (size_t)b * T_ * E_ + e;
    for (int t = 0; t < T_; ++t) acc += sc[t] * eb[(size_t)t * E_];  // coalesced
    ctx_bf[b * E_ + e] = (__bf16)(acc * inv);
  }
}

// G1[32,3H] = ctx @ Wc^T + xgi[:,t,:] ; G2[32,3H] = h @ W_hh^T + b_hh
// 384 waves (which x mt x 96 nt), unblocked: recurrence is latency-bound.
__global__ void gates_kernel(const __bf16* __restrict__ ctx_bf,
                             const __bf16* __restrict__ h_bf,
                             const __bf16* __restrict__ wihc_bf,
                             const __bf16* __restrict__ whh_bf,
                             const float* __restrict__ xgi,
                             const float* __restrict__ b_hh,
                             float* __restrict__ G1, float* __restrict__ G2,
                             int t) {
  int wid = blockIdx.x * (blockDim.x >> 5) + (threadIdx.x >> 5);
  int which = wid & 1;
  int mt = (wid >> 1) & 1;
  int nt = wid >> 2;
  const __bf16* A = which ? h_bf : ctx_bf;
  const __bf16* W = which ? whh_bf : wihc_bf;  // both compact, ld = 512
  v8f acc = {};
  for (int k0 = 0; k0 < H_; k0 += 32) {
    v16bf a = load_a_bf(A, H_, mt * 16, k0);
    v16bf b = load_b_bf(W, H_, nt * 16, k0, H3_);  // always in-bounds, no branch
    acc = wmma_bf16(a, b, acc);
  }
  int lane = threadIdx.x & 31;
  int n = nt * 16 + (lane & 15);
  int mb = mt * 16 + ((lane >> 4) << 3);
  if (which) {
    float bias = b_hh[n];
#pragma unroll
    for (int r = 0; r < 8; ++r) G2[(size_t)(mb + r) * H3_ + n] = acc[r] + bias;
  } else {
#pragma unroll
    for (int r = 0; r < 8; ++r)
      G1[(size_t)(mb + r) * H3_ + n] =
          acc[r] + xgi[((size_t)(mb + r) * L_ + t) * H3_ + n];
  }
}

// GRU elementwise: writes h (fp32, for attention), h_bf and hs_bf (GEMM inputs)
__global__ void gru_kernel(const float* __restrict__ G1,
                           const float* __restrict__ G2, float* __restrict__ h,
                           __bf16* __restrict__ h_bf, __bf16* __restrict__ hs_bf,
                           int t) {
  int i = blockIdx.x * blockDim.x + threadIdx.x;  // B*H
  int b = i >> 9, j = i & (H_ - 1);
  size_t base = (size_t)b * H3_;
  float r = 1.f / (1.f + __expf(-(G1[base + j] + G2[base + j])));
  float z = 1.f / (1.f + __expf(-(G1[base + H_ + j] + G2[base + H_ + j])));
  float n = tanhf(G1[base + 2 * H_ + j] + r * G2[base + 2 * H_ + j]);
  float hn = (1.f - z) * n + z * h[i];
  h[i] = hn;
  h_bf[i] = (__bf16)hn;
  hs_bf[((size_t)b * L_ + t) * H_ + j] = (__bf16)hn;
}

// ---------------------------------------------------------------------------

extern "C" void kernel_launch(void* const* d_in, const int* in_sizes, int n_in,
                              void* d_out, int out_size, void* d_ws,
                              size_t ws_size, hipStream_t stream) {
  (void)in_sizes; (void)n_in; (void)out_size; (void)ws_size;
  const float*         enc     = (const float*)d_in[0];
  const unsigned char* mask    = (const unsigned char*)d_in[1];
  const int*           targets = (const int*)d_in[2];
  const float*         emb     = (const float*)d_in[3];
  const float*         W_attn  = (const float*)d_in[4];
  const float*         W_ih    = (const float*)d_in[5];
  const float*         W_hh    = (const float*)d_in[6];
  const float*         b_ih    = (const float*)d_in[7];
  const float*         b_hh    = (const float*)d_in[8];
  const float*         W_out   = (const float*)d_in[9];
  const float*         b_out   = (const float*)d_in[10];
  float* out = (float*)d_out;

  // ---- workspace carve-up (~112 MB) ----
  float* ws   = (float*)d_ws;
  float* proj = ws;                                  // 8,388,608 f32
  float* xgi  = proj + (size_t)B_ * T_ * H_;         // 9,830,400 f32
  float* h    = xgi + (size_t)B_ * L_ * H3_;         // 16,384 f32
  float* G1   = h + B_ * H_;                         // 49,152 f32
  float* G2   = G1 + B_ * H3_;                       // 49,152 f32
  __bf16* enc_bf   = (__bf16*)(G2 + B_ * H3_);       // 8,388,608
  __bf16* emb_bf   = enc_bf + (size_t)B_ * T_ * E_;  // 2,560,000
  __bf16* wattn_bf = emb_bf + (size_t)C_ * H_;       // 262,144
  __bf16* wihx_bf  = wattn_bf + H_ * E_;             // 786,432
  __bf16* wihc_bf  = wihx_bf + H3_ * H_;             // 786,432
  __bf16* whh_bf   = wihc_bf + H3_ * E_;             // 786,432
  __bf16* wout_bf  = whh_bf + H3_ * H_;              // 2,560,000
  __bf16* hs_bf    = wout_bf + (size_t)C_ * H_;      // 3,276,800
  __bf16* h_bf     = hs_bf + (size_t)B_ * L_ * H_;   // 16,384
  __bf16* ctx_bf   = h_bf + B_ * H_;                 // 16,384

  // one-time bf16 staging (~70 MB traffic ~ 3us @ 23.3 TB/s)
  conv_kernel<<<(B_ * T_ * E_ / 2 + 255) / 256, 256, 0, stream>>>(enc, enc_bf,
                                                                  B_ * T_ * E_);
  conv_kernel<<<(C_ * H_ / 2 + 255) / 256, 256, 0, stream>>>(emb, emb_bf, C_ * H_);
  conv_kernel<<<(H_ * E_ / 2 + 255) / 256, 256, 0, stream>>>(W_attn, wattn_bf,
                                                             H_ * E_);
  conv_kernel<<<(H3_ * H_ / 2 + 255) / 256, 256, 0, stream>>>(W_hh, whh_bf,
                                                              H3_ * H_);
  conv_kernel<<<(C_ * H_ / 2 + 255) / 256, 256, 0, stream>>>(W_out, wout_bf,
                                                             C_ * H_);
  conv_wih_kernel<<<(H3_ * (H_ + E_)) / 256, 256, 0, stream>>>(W_ih, wihx_bf,
                                                               wihc_bf);
  init_h_kernel<<<(B_ * H_) / 256, 256, 0, stream>>>(h, h_bf);

  // big GEMMs: 8 waves/block, each wave a 32x64 strip; block shares B columns
  proj_kernel<<<64 * 8, 256, 0, stream>>>(enc_bf, wattn_bf, proj);
  xgi_kernel<<<25 * 24, 256, 0, stream>>>(targets, emb_bf, wihx_bf, b_ih, xgi);

  // sequential recurrence; stream order carries the h dependency
  for (int t = 0; t < L_; ++t) {
    attn_kernel<<<B_, 256, 0, stream>>>(proj, enc, mask, h, ctx_bf);
    gates_kernel<<<48, 256, 0, stream>>>(ctx_bf, h_bf, wihc_bf, whh_bf, xgi, b_hh,
                                         G1, G2, t);
    gru_kernel<<<(B_ * H_) / 256, 256, 0, stream>>>(G1, G2, h, h_bf, hs_bf, t);
  }

  // output projection: 25*79 blocks, 32x64 strip per wave
  logits_kernel<<<25 * 79, 256, 0, stream>>>(hs_bf, wout_bf, b_out, out);
}
